// APPNP_Link_27152783245328
// MI455X (gfx1250) — matrix operand order repaired and checked by reading, not compiled, not run on hardware
//
#include <hip/hip_runtime.h>
#include <math.h>

typedef __attribute__((ext_vector_type(16))) _Float16 v16h;
typedef __attribute__((ext_vector_type(8)))  float    v8f;

#define FN 128          // feature width (both layers)
#define KSTEPS 10
#define ALPHA 0.1f

// ---------------------------------------------------------------------------
// CSR build: in-degree histogram, exclusive scan, edge placement
// ---------------------------------------------------------------------------
__global__ __launch_bounds__(256) void zero_int_kernel(int* p, int n) {
  int i = blockIdx.x * blockDim.x + threadIdx.x;
  if (i < n) p[i] = 0;
}

__global__ __launch_bounds__(256) void count_indeg_kernel(const int* __restrict__ dst,
                                                          int* indeg, int e) {
  int i = blockIdx.x * blockDim.x + threadIdx.x;
  if (i < e) atomicAdd(&indeg[dst[i]], 1);
}

// dinv[i] = rsqrt(indeg[i] + 1)   (self-loop included; always > 0)
__global__ __launch_bounds__(256) void dinv_kernel(const int* __restrict__ indeg,
                                                   float* __restrict__ dinv, int n) {
  int i = blockIdx.x * blockDim.x + threadIdx.x;
  if (i < n) dinv[i] = rsqrtf((float)(indeg[i] + 1));
}

// single-block exclusive scan (exact integer, deterministic), 1024 threads
__global__ __launch_bounds__(1024) void scan_kernel(const int* __restrict__ cnt,
                                                    int* __restrict__ rowstart, int n) {
  __shared__ int lds[1024];
  __shared__ int carry;
  const int tid = threadIdx.x;
  if (tid == 0) carry = 0;
  __syncthreads();
  for (int base = 0; base < n; base += 1024) {
    int i = base + tid;
    int v = (i < n) ? cnt[i] : 0;
    lds[tid] = v;
    __syncthreads();
    for (int off = 1; off < 1024; off <<= 1) {
      int t = (tid >= off) ? lds[tid - off] : 0;
      __syncthreads();
      lds[tid] += t;
      __syncthreads();
    }
    int incl = lds[tid];
    int c = carry;
    if (i < n) rowstart[i] = c + incl - v;       // exclusive
    int total = lds[1023];
    __syncthreads();
    if (tid == 0) carry = c + total;
    __syncthreads();
  }
  if (tid == 0) rowstart[n] = carry;
}

// place edges into CSR order, precompute w = dinv[src]*dinv[dst]
__global__ __launch_bounds__(256) void place_kernel(const int* __restrict__ src,
                                                    const int* __restrict__ dst,
                                                    const float* __restrict__ dinv,
                                                    const int* __restrict__ rowstart,
                                                    int* __restrict__ cursor,
                                                    int* __restrict__ src_s,
                                                    float* __restrict__ w_s, int e) {
  int i = blockIdx.x * blockDim.x + threadIdx.x;
  if (i < e) {
    int s = src[i], d = dst[i];
    int pos = rowstart[d] + atomicAdd(&cursor[d], 1);
    src_s[pos] = s;
    w_s[pos]   = dinv[s] * dinv[d];
  }
}

// ---------------------------------------------------------------------------
// GEMM: C[nrows,128] = A[nrows,128] @ W^T + bias, W is [128,128] row-major.
// One wave per 16x16 tile, f16 WMMA, f32 accumulate. nrows % 16 == 0.
// ---------------------------------------------------------------------------
__global__ __launch_bounds__(256) void gemm_bias_wmma(const float* __restrict__ A,
                                                      const float* __restrict__ W,
                                                      const float* __restrict__ bias,
                                                      float* __restrict__ C, int nrows) {
  const int lane = threadIdx.x & 31;
  const int wave = blockIdx.x * 8 + (threadIdx.x >> 5);
  const int ntn  = FN / 16;                       // 8 column tiles
  const int tm   = wave / ntn;
  const int tn   = wave % ntn;
  if (tm * 16 >= nrows) return;                   // wave-uniform: EXEC stays all-1s

  const int half = lane >> 4;
  const int l16  = lane & 15;
  const int m    = tm * 16 + l16;
  const int n    = tn * 16 + l16;

  v8f c;
  const float bv = bias[n];
#pragma unroll
  for (int r = 0; r < 8; ++r) c[r] = bv;

  for (int kc = 0; kc < FN; kc += 32) {
    const float* arow = A + (size_t)m * FN + kc + 8 * half;
    v16h a;
#pragma unroll
    for (int j = 0; j < 8; ++j) {
      a[j]     = (_Float16)arow[j];               // VGPR0-3: K = 8*half + j
      a[8 + j] = (_Float16)arow[16 + j];          // VGPR4-7: K = 16 + 8*half + j
    }
    const float* wrow = W + (size_t)n * FN + kc + 8 * half;
    v16h b;
#pragma unroll
    for (int j = 0; j < 8; ++j) {
      b[j]     = (_Float16)wrow[j];
      b[8 + j] = (_Float16)wrow[16 + j];
    }

    c = __builtin_amdgcn_wmma_f32_16x16x32_f16(false, a, false, b,
                                               (short)0, c, false, false);
  }

  float* crow = C + (size_t)(tm * 16 + 8 * half) * FN + tn * 16 + l16;
#pragma unroll
  for (int r = 0; r < 8; ++r) crow[(size_t)r * FN] = c[r];
}

// ---------------------------------------------------------------------------
// Fused APPNP step (gather, no atomics):
//   Hout[i] = (relu?) 0.9*( dinv[i]^2*Hin[i] + sum_j w_j*Hin[src_j] ) + 0.1*X0[i]
// One wave per node; 32 lanes x float4 covers the 128-wide row.
// ---------------------------------------------------------------------------
__global__ __launch_bounds__(256) void prop_kernel(const float* __restrict__ Hin,
                                                   const float* __restrict__ X0,
                                                   const float* __restrict__ dinv,
                                                   const int* __restrict__ rowstart,
                                                   const int* __restrict__ src_s,
                                                   const float* __restrict__ w_s,
                                                   float* __restrict__ Hout,
                                                   int n, int relu) {
  int node = blockIdx.x * 8 + (threadIdx.x >> 5);
  if (node >= n) return;
  int lane = threadIdx.x & 31;

  float di = dinv[node];
  float d2 = di * di;
  float4 hv = ((const float4*)(Hin + (size_t)node * FN))[lane];
  float4 acc;
  acc.x = d2 * hv.x; acc.y = d2 * hv.y; acc.z = d2 * hv.z; acc.w = d2 * hv.w;

  int j  = rowstart[node];
  int j1 = rowstart[node + 1];
  // software pipeline: prefetch next edge's (src, w) while row load is in flight
  int   sN = 0; float wN = 0.0f;
  if (j < j1) { sN = src_s[j]; wN = w_s[j]; }
  while (j < j1) {
    int   s = sN;
    float w = wN;
    ++j;
    if (j < j1) { sN = src_s[j]; wN = w_s[j]; }
    float4 v = ((const float4*)(Hin + (size_t)s * FN))[lane];
    acc.x += w * v.x; acc.y += w * v.y; acc.z += w * v.z; acc.w += w * v.w;
  }

  float4 x = ((const float4*)(X0 + (size_t)node * FN))[lane];
  float4 h;
  h.x = (1.0f - ALPHA) * acc.x + ALPHA * x.x;
  h.y = (1.0f - ALPHA) * acc.y + ALPHA * x.y;
  h.z = (1.0f - ALPHA) * acc.z + ALPHA * x.z;
  h.w = (1.0f - ALPHA) * acc.w + ALPHA * x.w;
  if (relu) {
    h.x = fmaxf(h.x, 0.0f); h.y = fmaxf(h.y, 0.0f);
    h.z = fmaxf(h.z, 0.0f); h.w = fmaxf(h.w, 0.0f);
  }
  ((float4*)(Hout + (size_t)node * FN))[lane] = h;
}

// ---------------------------------------------------------------------------
// Link head: out[p] = log_softmax( [h[i0], h[i1]] @ W3^T + b3 ), one wave/pair
// ---------------------------------------------------------------------------
__global__ __launch_bounds__(256) void link_out_kernel(const float* __restrict__ H,
                                                       const int* __restrict__ index,
                                                       const float* __restrict__ W3,
                                                       const float* __restrict__ b3,
                                                       float* __restrict__ out, int p) {
  int wave = blockIdx.x * 8 + (threadIdx.x >> 5);
  if (wave >= p) return;
  int lane = threadIdx.x & 31;
  int i0 = index[wave * 2 + 0];
  int i1 = index[wave * 2 + 1];
  float4 a  = ((const float4*)(H + (size_t)i0 * FN))[lane];
  float4 bb = ((const float4*)(H + (size_t)i1 * FN))[lane];

  float o[2];
#pragma unroll
  for (int jj = 0; jj < 2; ++jj) {
    float4 w0 = ((const float4*)(W3 + jj * 2 * FN))[lane];        // cols 0..127
    float4 w1 = ((const float4*)(W3 + jj * 2 * FN + FN))[lane];   // cols 128..255
    float s = a.x * w0.x + a.y * w0.y + a.z * w0.z + a.w * w0.w
            + bb.x * w1.x + bb.y * w1.y + bb.z * w1.z + bb.w * w1.w;
#pragma unroll
    for (int off = 16; off > 0; off >>= 1) s += __shfl_xor(s, off, 32);
    o[jj] = s + b3[jj];
  }
  if (lane == 0) {
    float m   = fmaxf(o[0], o[1]);
    float lse = m + logf(expf(o[0] - m) + expf(o[1] - m));
    out[wave * 2 + 0] = o[0] - lse;
    out[wave * 2 + 1] = o[1] - lse;
  }
}

// ---------------------------------------------------------------------------
extern "C" void kernel_launch(void* const* d_in, const int* in_sizes, int n_in,
                              void* d_out, int out_size, void* d_ws, size_t ws_size,
                              hipStream_t stream) {
  (void)n_in; (void)out_size; (void)ws_size;
  const float* x   = (const float*)d_in[0];
  const int*   ei  = (const int*)d_in[1];     // [2, E]
  const int*   idx = (const int*)d_in[2];     // [P, 2]
  const float* W1  = (const float*)d_in[3];
  const float* b1  = (const float*)d_in[4];
  const float* W2  = (const float*)d_in[5];
  const float* b2  = (const float*)d_in[6];
  const float* W3  = (const float*)d_in[7];
  const float* b3  = (const float*)d_in[8];
  float* out = (float*)d_out;

  const int N = in_sizes[0] / FN;             // 50000
  const int E = in_sizes[1] / 2;              // 800000
  const int P = in_sizes[2] / 2;              // 200000
  const int* src = ei;
  const int* dst = ei + E;

  // workspace layout (256B aligned)
  size_t off = 0;
  auto alloc = [&](size_t bytes) {
    void* ptr = (char*)d_ws + off;
    off += (bytes + 255) & ~(size_t)255;
    return ptr;
  };
  float* H0       = (float*)alloc((size_t)N * FN * sizeof(float));   // x0 of current round
  float* Ha       = (float*)alloc((size_t)N * FN * sizeof(float));
  float* Hb       = (float*)alloc((size_t)N * FN * sizeof(float));
  float* dinv     = (float*)alloc((size_t)N * sizeof(float));
  int*   indeg    = (int*)  alloc((size_t)N * sizeof(int));
  int*   rowstart = (int*)  alloc((size_t)(N + 1) * sizeof(int));
  int*   cursor   = (int*)  alloc((size_t)N * sizeof(int));
  int*   src_s    = (int*)  alloc((size_t)E * sizeof(int));
  float* w_s      = (float*)alloc((size_t)E * sizeof(float));

  const dim3 blk(256);
  const int gN  = (N + 255) / 256;
  const int gE  = (E + 255) / 256;
  const int gNw = (N + 7) / 8;                // 8 waves / block, wave per node
  const int gPw = (P + 7) / 8;
  const int gG  = ((N / 16) * (FN / 16) + 7) / 8;

  // ---- CSR build + GCN normalization (one time) ----
  zero_int_kernel<<<gN, blk, 0, stream>>>(indeg, N);
  zero_int_kernel<<<gN, blk, 0, stream>>>(cursor, N);
  count_indeg_kernel<<<gE, blk, 0, stream>>>(dst, indeg, E);
  dinv_kernel<<<gN, blk, 0, stream>>>(indeg, dinv, N);
  scan_kernel<<<1, 1024, 0, stream>>>(indeg, rowstart, N);
  place_kernel<<<gE, blk, 0, stream>>>(src, dst, dinv, rowstart, cursor, src_s, w_s, E);

  // ---- round 1: H0 = x @ W1^T + b1 ; APPNP(10) ; relu ----
  gemm_bias_wmma<<<gG, blk, 0, stream>>>(x, W1, b1, H0, N);
  {
    const float* hin = H0;
    float* hout = Ha;
    for (int k = 0; k < KSTEPS; ++k) {
      prop_kernel<<<gNw, blk, 0, stream>>>(hin, H0, dinv, rowstart, src_s, w_s,
                                           hout, N, (k == KSTEPS - 1) ? 1 : 0);
      hin  = hout;
      hout = (hout == Ha) ? Hb : Ha;
    }
  }
  // after 10 steps starting H0->Ha, result lives in Hb
  // ---- round 2: H0 = Hb @ W2^T + b2 ; APPNP(10) ; relu ----
  gemm_bias_wmma<<<gG, blk, 0, stream>>>(Hb, W2, b2, H0, N);
  {
    const float* hin = H0;
    float* hout = Ha;
    for (int k = 0; k < KSTEPS; ++k) {
      prop_kernel<<<gNw, blk, 0, stream>>>(hin, H0, dinv, rowstart, src_s, w_s,
                                           hout, N, (k == KSTEPS - 1) ? 1 : 0);
      hin  = hout;
      hout = (hout == Ha) ? Hb : Ha;
    }
  }

  // ---- link prediction head (final features in Hb) ----
  link_out_kernel<<<gPw, blk, 0, stream>>>(Hb, idx, W3, b3, out, P);
}
